// xGINE_16028817949316
// MI455X (gfx1250) — matrix operand order repaired and checked by reading, not compiled, not run on hardware
//
#include <hip/hip_runtime.h>
#include <hip/hip_bf16.h>

typedef __attribute__((ext_vector_type(16))) __bf16 v16bf;
typedef __attribute__((ext_vector_type(8)))  float  v8f;

#define A_STRIDE 80   // bf16 elements per LDS row (64 data + 16 pad, keeps 16B alignment)

// ---------------------------------------------------------------- utilities
__global__ void zero_f32(float* __restrict__ p, int n) {
    int t = blockIdx.x * blockDim.x + threadIdx.x;
    if (t < n) p[t] = 0.0f;
}

__global__ void zero_i32(int* __restrict__ p, int n) {
    int t = blockIdx.x * blockDim.x + threadIdx.x;
    if (t < n) p[t] = 0;
}

// Pack a 64x64 fp32 weight W (row-major, out x in; used as h @ W^T) into
// per-lane WMMA B-fragment order: dst[((kt*4+nt)*32+lane)*16 + j] = W[n][k]
//   n = nt*16 + (lane&15),  k = kt*32 + (lane>>4)*8 + (j&7) + (j>>3)*16
__global__ void pack_weight_bf16(const float* __restrict__ W, __bf16* __restrict__ dst) {
    int idx = blockIdx.x * blockDim.x + threadIdx.x;
    if (idx >= 4096) return;
    int j    = idx & 15;
    int lane = (idx >> 4) & 31;
    int nt   = (idx >> 9) & 3;
    int kt   = (idx >> 11) & 1;
    int n = nt * 16 + (lane & 15);
    int k = kt * 32 + ((lane >> 4) << 3) + (j & 7) + ((j >> 3) << 4);
    dst[idx] = (__bf16)W[n * 64 + k];
}

// ---------------------------------------------------------------- CSR build (by destination)
__global__ void hist_kernel(const int* __restrict__ dst, int* __restrict__ deg, int n_edges) {
    int e = blockIdx.x * blockDim.x + threadIdx.x;
    if (e < n_edges) atomicAdd(&deg[dst[e]], 1);
}

// per-1024-block inclusive scan -> exclusive start[], block totals -> bsum[]
__global__ void __launch_bounds__(1024)
scan_block_kernel(const int* __restrict__ deg, int* __restrict__ start,
                  int* __restrict__ bsum, int n)
{
    __shared__ int buf[1024];
    int i = blockIdx.x * 1024 + threadIdx.x;
    int v = (i < n) ? deg[i] : 0;
    buf[threadIdx.x] = v;
    __syncthreads();
    for (int off = 1; off < 1024; off <<= 1) {
        int t = (threadIdx.x >= off) ? buf[threadIdx.x - off] : 0;
        __syncthreads();
        buf[threadIdx.x] += t;
        __syncthreads();
    }
    if (i < n) start[i] = buf[threadIdx.x] - v;   // exclusive prefix
    if (threadIdx.x == 1023) bsum[blockIdx.x] = buf[1023];
}

__global__ void scan_serial_kernel(int* __restrict__ bsum, int nb) {
    if (threadIdx.x == 0 && blockIdx.x == 0) {
        int run = 0;
        for (int b = 0; b < nb; ++b) { int t = bsum[b]; bsum[b] = run; run += t; }
    }
}

__global__ void scan_add_kernel(int* __restrict__ start, const int* __restrict__ bsum, int n) {
    int i = blockIdx.x * blockDim.x + threadIdx.x;
    if (i < n) start[i] += bsum[i >> 10];
}

__global__ void fill_kernel(const int* __restrict__ dst, const int* __restrict__ start,
                            int* __restrict__ cursor, int* __restrict__ eids, int n_edges)
{
    int e = blockIdx.x * blockDim.x + threadIdx.x;
    if (e < n_edges) {
        int d = dst[e];
        int p = atomicAdd(&cursor[d], 1);
        eids[start[d] + p] = e;
    }
}

// ---------------------------------------------------------------- GINE aggregation (gather, atomic-free)
// One wave per node; node id is wave-uniform (readfirstlane) so the edge-list
// walk (eids/src/eattr) compiles to scalar loads; each lane owns 2 channels.
// agg[node][c] = sum_e relu(h[src(e)][c] + eattr[e]*ew[c] + eb[c])
__global__ void __launch_bounds__(256)
gine_gather_kernel(const float* __restrict__ h, const int* __restrict__ src,
                   const float* __restrict__ eattr,
                   const int* __restrict__ rstart, const int* __restrict__ rdeg,
                   const int* __restrict__ eids,
                   const float* __restrict__ ew, const float* __restrict__ eb,
                   float* __restrict__ agg, int n_nodes)
{
    int node = __builtin_amdgcn_readfirstlane((int)(blockIdx.x * 8 + (threadIdx.x >> 5)));
    if (node >= n_nodes) return;
    int c = (threadIdx.x & 31) << 1;

    float2 wv = *(const float2*)(ew + c);
    float2 bv = *(const float2*)(eb + c);
    float ax = 0.0f, ay = 0.0f;

    int s0 = rstart[node];
    int dn = rdeg[node];
    for (int i = 0; i < dn; ++i) {
        int e   = eids[s0 + i];
        int s   = src[e];
        float a = eattr[e];
        float2 hv = *(const float2*)(h + (size_t)s * 64 + c);
        ax += fmaxf(fmaf(a, wv.x, hv.x + bv.x), 0.0f);
        ay += fmaxf(fmaf(a, wv.y, hv.y + bv.y), 0.0f);
    }
    float2 o; o.x = ax; o.y = ay;
    *(float2*)(agg + (size_t)node * 64 + c) = o;
}

// ---------------------------------------------------------------- WMMA fragment loads
__device__ __forceinline__ v16bf load_a_frag(const __bf16* As, int lane, int mt, int kt) {
    // A 16x32 bf16: lanes 0-15 -> rows 0-15 with K {0..7, 16..23};
    //               lanes 16-31 -> rows 0-15 with K {8..15, 24..31}
    int row = mt * 16 + (lane & 15);
    int k0  = kt * 32 + ((lane >> 4) << 3);
    union { v16bf v; uint4 q[2]; } u;
    u.q[0] = *(const uint4*)(As + row * A_STRIDE + k0);
    u.q[1] = *(const uint4*)(As + row * A_STRIDE + k0 + 16);
    return u.v;
}

__device__ __forceinline__ v16bf load_b_frag(const __bf16* Bpack, int lane, int nt, int kt) {
    union { v16bf v; uint4 q[2]; } u;
    const uint4* p = (const uint4*)(Bpack + ((((kt << 2) + nt) * 32 + lane) << 4));
    u.q[0] = p[0];
    u.q[1] = p[1];
    return u.v;
}

// ---------------------------------------------------------------- fused node-update + MLP + BN stats
__global__ void __launch_bounds__(256)
mlp_kernel(const float* __restrict__ hin, const float* __restrict__ agg,
           const float* __restrict__ epsp,
           const __bf16* __restrict__ W1p, const float* __restrict__ b1,
           const __bf16* __restrict__ W2p, const float* __restrict__ b2,
           float* __restrict__ hout,
           float* __restrict__ bn_sum, float* __restrict__ bn_sumsq,
           int n_nodes)
{
    __shared__ __align__(16) __bf16 As[64 * A_STRIDE];
    __shared__ __align__(16) __bf16 Hs[64 * A_STRIDE];
    __shared__ float s_sum[64];
    __shared__ float s_ss[64];

    int tid = threadIdx.x;
    int rowbase = blockIdx.x * 64;
    float ep = 1.0f + epsp[0];

    // Stage xin = (1+eps)*h + agg into LDS as bf16 (packed 8B stores)
    for (int pass = 0; pass < 4; ++pass) {
        int row = (tid >> 4) + pass * 16;
        int col = (tid & 15) << 2;
        int node = rowbase + row;
        float4 v = make_float4(0.f, 0.f, 0.f, 0.f);
        if (node < n_nodes) {
            float4 hv = *(const float4*)(hin + (size_t)node * 64 + col);
            float4 av = *(const float4*)(agg + (size_t)node * 64 + col);
            v.x = fmaf(ep, hv.x, av.x);
            v.y = fmaf(ep, hv.y, av.y);
            v.z = fmaf(ep, hv.z, av.z);
            v.w = fmaf(ep, hv.w, av.w);
        }
        union { __bf16 b[4]; uint2 u2; } pk;
        pk.b[0] = (__bf16)v.x; pk.b[1] = (__bf16)v.y;
        pk.b[2] = (__bf16)v.z; pk.b[3] = (__bf16)v.w;
        *(uint2*)(As + row * A_STRIDE + col) = pk.u2;
    }
    if (tid < 64) { s_sum[tid] = 0.0f; s_ss[tid] = 0.0f; }
    __syncthreads();

    int lane = tid & 31;
    int wave = tid >> 5;

    // GEMM1 + bias + ReLU -> Hs (bf16)
    for (int t = wave; t < 16; t += 8) {
        int mt = t >> 2, nt = t & 3;
        v8f acc = {};
        for (int kt = 0; kt < 2; ++kt) {
            v16bf a = load_a_frag(As, lane, mt, kt);
            v16bf b = load_b_frag(W1p, lane, nt, kt);
            acc = __builtin_amdgcn_wmma_f32_16x16x32_bf16(
                false, a, false, b, (short)0, acc, false, false);
        }
        int col = nt * 16 + (lane & 15);
        int r0  = mt * 16 + ((lane >> 4) << 3);
        float bias = b1[col];
        for (int r = 0; r < 8; ++r) {
            float v = acc[r] + bias;
            v = v > 0.0f ? v : 0.0f;
            Hs[(r0 + r) * A_STRIDE + col] = (__bf16)v;
        }
    }
    __syncthreads();

    // GEMM2 + bias -> global; BN partial stats
    for (int t = wave; t < 16; t += 8) {
        int mt = t >> 2, nt = t & 3;
        v8f acc = {};
        for (int kt = 0; kt < 2; ++kt) {
            v16bf a = load_a_frag(Hs, lane, mt, kt);
            v16bf b = load_b_frag(W2p, lane, nt, kt);
            acc = __builtin_amdgcn_wmma_f32_16x16x32_bf16(
                false, a, false, b, (short)0, acc, false, false);
        }
        int col = nt * 16 + (lane & 15);
        int r0  = mt * 16 + ((lane >> 4) << 3);
        float bias = b2[col];
        float s = 0.0f, ss = 0.0f;
        for (int r = 0; r < 8; ++r) {
            int node = rowbase + r0 + r;
            float v = acc[r] + bias;
            if (node < n_nodes) {
                hout[(size_t)node * 64 + col] = v;
                s += v;
                ss += v * v;
            }
        }
        atomicAdd(&s_sum[col], s);
        atomicAdd(&s_ss[col], ss);
    }
    __syncthreads();

    if (tid < 64) {
        atomicAdd(&bn_sum[tid], s_sum[tid]);
        atomicAdd(&bn_sumsq[tid], s_ss[tid]);
    }
}

// ---------------------------------------------------------------- BN
__global__ void bn_finalize(const float* __restrict__ sum, const float* __restrict__ sumsq,
                            const float* __restrict__ gamma, const float* __restrict__ beta,
                            float* __restrict__ scale, float* __restrict__ shift, float invN)
{
    int c = threadIdx.x;
    if (c >= 64) return;
    float mu  = sum[c] * invN;
    float var = sumsq[c] * invN - mu * mu;
    float sc  = gamma[c] * rsqrtf(var + 1e-5f);
    scale[c] = sc;
    shift[c] = beta[c] - mu * sc;
}

__global__ void bn_apply_relu(const float* __restrict__ hin,
                              const float* __restrict__ scale, const float* __restrict__ shift,
                              float* __restrict__ hout, int total)
{
    int t = blockIdx.x * blockDim.x + threadIdx.x;
    if (t >= total) return;
    int c = t & 63;
    float v = fmaf(hin[t], scale[c], shift[c]);
    hout[t] = v > 0.0f ? v : 0.0f;
}

// ---------------------------------------------------------------- pooling + final linear
__global__ void pool_sum_kernel(const float* __restrict__ h, const int* __restrict__ batch,
                                float* __restrict__ pooled, int n_nodes)
{
    int t = blockIdx.x * blockDim.x + threadIdx.x;
    if (t >= n_nodes * 64) return;
    int node = t >> 6, c = t & 63;
    atomicAdd(&pooled[batch[node] * 64 + c], h[t]);
}

__global__ void pool_cnt_kernel(const int* __restrict__ batch, float* __restrict__ cnts, int n_nodes)
{
    int node = blockIdx.x * blockDim.x + threadIdx.x;
    if (node < n_nodes) atomicAdd(&cnts[batch[node]], 1.0f);
}

__global__ void final_linear_kernel(const float* __restrict__ pooled, const float* __restrict__ cnts,
                                    const float* __restrict__ lw, const float* __restrict__ lb,
                                    float* __restrict__ out)
{
    int t = threadIdx.x;             // 640 = 64 graphs x 10 outputs
    if (t >= 640) return;
    int g = t / 10, o = t - g * 10;
    float inv = 1.0f / fmaxf(cnts[g], 1.0f);
    float acc = lb[o];
    for (int c = 0; c < 64; ++c)
        acc += pooled[g * 64 + c] * inv * lw[o * 64 + c];
    out[t] = acc;
}

// ---------------------------------------------------------------- host
extern "C" void kernel_launch(void* const* d_in, const int* in_sizes, int n_in,
                              void* d_out, int out_size, void* d_ws, size_t ws_size,
                              hipStream_t stream)
{
    const float* x        = (const float*)d_in[0];
    const int*   eidx     = (const int*)d_in[1];
    const int*   batch    = (const int*)d_in[2];
    const float* eattr    = (const float*)d_in[3];
    const float* lin_e0_w = (const float*)d_in[4];
    const float* lin_e0_b = (const float*)d_in[5];
    const float* mlp0_w1  = (const float*)d_in[6];
    const float* mlp0_b1  = (const float*)d_in[7];
    const float* mlp0_w2  = (const float*)d_in[8];
    const float* mlp0_b2  = (const float*)d_in[9];
    const float* eps0     = (const float*)d_in[10];
    const float* lin_e1_w = (const float*)d_in[11];
    const float* lin_e1_b = (const float*)d_in[12];
    const float* mlp1_w1  = (const float*)d_in[13];
    const float* mlp1_b1  = (const float*)d_in[14];
    const float* mlp1_w2  = (const float*)d_in[15];
    const float* mlp1_b2  = (const float*)d_in[16];
    const float* eps1     = (const float*)d_in[17];
    const float* bn_gamma = (const float*)d_in[18];
    const float* bn_beta  = (const float*)d_in[19];
    const float* lin_w    = (const float*)d_in[20];
    const float* lin_b    = (const float*)d_in[21];

    const int N = in_sizes[0] / 64;     // 100000 nodes
    const int E = in_sizes[3];          // 1600000 edges
    const int NG = 64;

    // workspace carve-up
    char* ws = (char*)d_ws;
    size_t off = 0;
    auto carve = [&](size_t bytes) -> char* {
        char* p = ws + off;
        off = (off + bytes + 255) & ~(size_t)255;
        return p;
    };
    float*  h_a    = (float*)carve((size_t)N * 64 * sizeof(float));
    float*  h_b    = (float*)carve((size_t)N * 64 * sizeof(float));
    float*  agg    = (float*)carve((size_t)N * 64 * sizeof(float));
    int*    deg    = (int*)carve((size_t)N * sizeof(int));
    int*    rstart = (int*)carve((size_t)N * sizeof(int));
    int*    cursor = (int*)carve((size_t)N * sizeof(int));
    int*    eids   = (int*)carve((size_t)E * sizeof(int));
    int*    bsum   = (int*)carve(512 * sizeof(int));
    __bf16* W1p0   = (__bf16*)carve(4096 * sizeof(__bf16));
    __bf16* W2p0   = (__bf16*)carve(4096 * sizeof(__bf16));
    __bf16* W1p1   = (__bf16*)carve(4096 * sizeof(__bf16));
    __bf16* W2p1   = (__bf16*)carve(4096 * sizeof(__bf16));
    float*  bn_sum = (float*)carve(64 * sizeof(float));
    float*  bn_ss  = (float*)carve(64 * sizeof(float));
    float*  bscale = (float*)carve(64 * sizeof(float));
    float*  bshift = (float*)carve(64 * sizeof(float));
    float*  pooled = (float*)carve(NG * 64 * sizeof(float));
    float*  cnts   = (float*)carve(NG * sizeof(float));

    const int* esrc = eidx;
    const int* edst = eidx + E;

    const int feat_total  = N * 64;
    const int feat_blocks = (feat_total + 255) / 256;
    const int node_blocks = (N + 255) / 256;
    const int edge_blocks = (E + 255) / 256;
    const int scan_blocks = (N + 1023) / 1024;
    const int mlp_blocks  = (N + 63) / 64;
    const int gath_blocks = (N + 7) / 8;
    const float invN = 1.0f / (float)N;

    // pack weights to WMMA B-fragment bf16 layout
    pack_weight_bf16<<<16, 256, 0, stream>>>(mlp0_w1, W1p0);
    pack_weight_bf16<<<16, 256, 0, stream>>>(mlp0_w2, W2p0);
    pack_weight_bf16<<<16, 256, 0, stream>>>(mlp1_w1, W1p1);
    pack_weight_bf16<<<16, 256, 0, stream>>>(mlp1_w2, W2p1);

    // ---------------- build CSR by destination (reused for all 3 layers) ----
    zero_i32<<<node_blocks, 256, 0, stream>>>(deg, N);
    zero_i32<<<node_blocks, 256, 0, stream>>>(cursor, N);
    hist_kernel<<<edge_blocks, 256, 0, stream>>>(edst, deg, E);
    scan_block_kernel<<<scan_blocks, 1024, 0, stream>>>(deg, rstart, bsum, N);
    scan_serial_kernel<<<1, 1, 0, stream>>>(bsum, scan_blocks);
    scan_add_kernel<<<node_blocks, 256, 0, stream>>>(rstart, bsum, N);
    fill_kernel<<<edge_blocks, 256, 0, stream>>>(edst, rstart, cursor, eids, E);

    // ---------------- layer 0 ----------------
    gine_gather_kernel<<<gath_blocks, 256, 0, stream>>>(x, esrc, eattr,
                                                        rstart, deg, eids,
                                                        lin_e0_w, lin_e0_b, agg, N);
    zero_f32<<<1, 128, 0, stream>>>(bn_sum, 128);   // bn_sum + bn_ss contiguous
    mlp_kernel<<<mlp_blocks, 256, 0, stream>>>(x, agg, eps0,
                                               W1p0, mlp0_b1, W2p0, mlp0_b2,
                                               h_b, bn_sum, bn_ss, N);
    bn_finalize<<<1, 64, 0, stream>>>(bn_sum, bn_ss, bn_gamma, bn_beta,
                                      bscale, bshift, invN);
    bn_apply_relu<<<feat_blocks, 256, 0, stream>>>(h_b, bscale, bshift, h_a, feat_total);

    // ---------------- layers 1..2 ----------------
    for (int L = 1; L < 3; ++L) {
        gine_gather_kernel<<<gath_blocks, 256, 0, stream>>>(h_a, esrc, eattr,
                                                            rstart, deg, eids,
                                                            lin_e1_w, lin_e1_b, agg, N);
        zero_f32<<<1, 128, 0, stream>>>(bn_sum, 128);
        mlp_kernel<<<mlp_blocks, 256, 0, stream>>>(h_a, agg, eps1,
                                                   W1p1, mlp1_b1, W2p1, mlp1_b2,
                                                   h_b, bn_sum, bn_ss, N);
        bn_finalize<<<1, 64, 0, stream>>>(bn_sum, bn_ss,
                                          bn_gamma + 64 * L, bn_beta + 64 * L,
                                          bscale, bshift, invN);
        bn_apply_relu<<<feat_blocks, 256, 0, stream>>>(h_b, bscale, bshift, h_a, feat_total);
    }

    // ---------------- pooling + final linear ----------------
    zero_f32<<<(NG * 64 + 255) / 256, 256, 0, stream>>>(pooled, NG * 64);
    zero_f32<<<1, 64, 0, stream>>>(cnts, NG);
    pool_sum_kernel<<<feat_blocks, 256, 0, stream>>>(h_a, batch, pooled, N);
    pool_cnt_kernel<<<node_blocks, 256, 0, stream>>>(batch, cnts, N);
    final_linear_kernel<<<1, 640, 0, stream>>>(pooled, cnts, lin_w, lin_b, (float*)d_out);
}
